// TemporalAttention_57200374448417
// MI455X (gfx1250) — matrix operand order repaired and checked by reading, not compiled, not run on hardware
//
#include <hip/hip_runtime.h>

typedef __attribute__((ext_vector_type(16))) __bf16 v16bf;
typedef __attribute__((ext_vector_type(8)))  float  v8f;

#define DMODEL 512
#define HEADS  8
#define HDIM   64
#define TSEQ   512

// ---------------- CDNA5 async global->LDS helpers ---------------------------
// GLOBAL_LOAD_ASYNC_TO_LDS_B128: per-lane LDS[vdst] = MEM[vaddr], ASYNCcnt.
__device__ __forceinline__ void async_b128(unsigned lds_off,
                                           unsigned long long gaddr) {
  asm volatile("global_load_async_to_lds_b128 %0, %1, off" ::"v"(lds_off),
               "v"(gaddr)
               : "memory");
}
__device__ __forceinline__ void wait_async0() {
  asm volatile("s_wait_asynccnt 0x0" ::: "memory");
}
__device__ __forceinline__ unsigned lds_addr(const void* p) {
  return (unsigned)(size_t)p;  // LDS flat address: offset lives in low 32 bits
}

// ---------------- WMMA fragment loaders (bf16, 16x16x32) --------------------
// A 16x32 (MxK): row m; lane<16 -> elems 0..7 = K0..7, 8..15 = K16..23
//                       lane>=16 -> elems 0..7 = K8..15, 8..15 = K24..31
__device__ __forceinline__ v16bf frag_a_ld(const __bf16* base, int lda, int m,
                                           int kk, int lane) {
  const __bf16* p = base + m * lda + kk + ((lane & 16) ? 8 : 0);
  v16bf r;
#pragma unroll
  for (int i = 0; i < 8; ++i) r[i] = p[i];
#pragma unroll
  for (int i = 0; i < 8; ++i) r[i + 8] = p[16 + i];
  return r;
}

// B 32x16 (KxN): N = nBase + lane%16, elements = K in same order as A (k-major LDS)
__device__ __forceinline__ v16bf frag_b_ld(const __bf16* base, int ldb,
                                           int kBase, int nBase, int lane) {
  const __bf16* p = base + (kBase + ((lane & 16) ? 8 : 0)) * ldb + nBase + (lane & 15);
  v16bf r;
#pragma unroll
  for (int i = 0; i < 8; ++i) r[i] = p[i * ldb];
#pragma unroll
  for (int i = 0; i < 8; ++i) r[i + 8] = p[(16 + i) * ldb];
  return r;
}

__device__ __forceinline__ float block_sum(float v, float* red) {
  int tid = threadIdx.x;
  red[tid] = v;
  __syncthreads();
  for (int s = 128; s > 0; s >>= 1) {
    if (tid < s) red[tid] += red[tid + s];
    __syncthreads();
  }
  float r = red[0];
  __syncthreads();
  return r;
}

// ---------------- kernel 1: X + timing signal, LayerNorm -> Xn fp32 + bf16 ---
__global__ __launch_bounds__(256) void prep_ln_kernel(
    const float* __restrict__ X, const float* __restrict__ g,
    const float* __restrict__ be, float* __restrict__ Xn,
    __bf16* __restrict__ Xnb) {
  __shared__ float red[256];
  int token = blockIdx.x;
  int t = token % TSEQ;
  int tid = threadIdx.x;
  const float log_inc = 9.210340371976184f / 255.0f;  // ln(1e4)/(256-1)
  float x[2];
#pragma unroll
  for (int i = 0; i < 2; ++i) {
    int c = tid + i * 256;
    int j = (c < 256) ? c : (c - 256);
    float sc = (float)t * __expf(-(float)j * log_inc);
    float sig = (c < 256) ? __sinf(sc) : __cosf(sc);
    x[i] = X[(size_t)token * DMODEL + c] + sig;
  }
  float mean = block_sum(x[0] + x[1], red) * (1.0f / DMODEL);
  float d0 = x[0] - mean, d1 = x[1] - mean;
  float var = block_sum(d0 * d0 + d1 * d1, red) * (1.0f / (DMODEL - 1));
  float denom = sqrtf(var) + 1e-6f;  // unbiased std + eps (matches reference)
#pragma unroll
  for (int i = 0; i < 2; ++i) {
    int c = tid + i * 256;
    float y = g[c] * (x[i] - mean) / denom + be[c];
    Xn[(size_t)token * DMODEL + c] = y;
    Xnb[(size_t)token * DMODEL + c] = (__bf16)y;
  }
}

// ---------------- WMMA GEMM: C = [relu](A(bf16)[M,K] @ W(f32)[N,K]^T + b) ----
// block tile 256x64, BK=32, double-buffered LDS, async A staging.
// 8 waves; wave w -> 32-row strip (2 A frags x 4 B frags = 8 WMMA / step).
template <bool RELU>
__global__ __launch_bounds__(256) void gemm_kernel(
    const __bf16* __restrict__ A, const float* __restrict__ W,
    const float* __restrict__ bias, float* __restrict__ outF,
    __bf16* __restrict__ outB, const float* __restrict__ residual, int Kdim,
    int Nout) {
  __shared__ __bf16 As[2][256 * 32];  // 2 x 16 KB
  __shared__ __bf16 Ws[2][32 * 64];   // 2 x 4 KB, k-major (transposed store)
  int tid = threadIdx.x;
  int lane = tid & 31, wave = tid >> 5;
  size_t mBlock = (size_t)blockIdx.x * 256;
  int nBlock = blockIdx.y * 64;

  const __bf16* aSrc = A + (mBlock + tid) * Kdim;  // one 64B row per thread
  int wN = tid >> 2, wK = (tid & 3) * 8;
  const float* wSrc = W + (size_t)(nBlock + wN) * Kdim + wK;
  float wreg[8];

  {  // prologue: stage k-panel 0 into buffer 0
    unsigned lds = lds_addr(&As[0][tid * 32]);
    unsigned long long g = (unsigned long long)(size_t)aSrc;
    async_b128(lds, g);
    async_b128(lds + 16, g + 16);
    async_b128(lds + 32, g + 32);
    async_b128(lds + 48, g + 48);
#pragma unroll
    for (int i = 0; i < 8; ++i) Ws[0][(wK + i) * 64 + wN] = (__bf16)wSrc[i];
  }
  wait_async0();
  __syncthreads();

  v8f acc[8] = {};
  int nStep = Kdim >> 5;
  int buf = 0;
  for (int s = 0; s < nStep; ++s) {
    if (s + 1 < nStep) {  // prefetch next panel into other buffer
      int k1 = (s + 1) << 5;
      unsigned lds = lds_addr(&As[buf ^ 1][tid * 32]);
      unsigned long long g = (unsigned long long)(size_t)(aSrc + k1);
      async_b128(lds, g);
      async_b128(lds + 16, g + 16);
      async_b128(lds + 32, g + 32);
      async_b128(lds + 48, g + 48);
#pragma unroll
      for (int i = 0; i < 8; ++i) wreg[i] = wSrc[k1 + i];
    }
    int m0 = wave * 32 + (lane & 15);
    v16bf a0 = frag_a_ld(As[buf], 32, m0, 0, lane);
    v16bf a1 = frag_a_ld(As[buf], 32, m0 + 16, 0, lane);
#pragma unroll
    for (int tt = 0; tt < 4; ++tt) {
      v16bf b = frag_b_ld(Ws[buf], 64, 0, tt * 16, lane);
      acc[tt] = __builtin_amdgcn_wmma_f32_16x16x32_bf16(false, a0, false, b,
                                                        (short)0, acc[tt],
                                                        false, false);
      acc[4 + tt] = __builtin_amdgcn_wmma_f32_16x16x32_bf16(
          false, a1, false, b, (short)0, acc[4 + tt], false, false);
    }
    if (s + 1 < nStep) {
#pragma unroll
      for (int i = 0; i < 8; ++i)
        Ws[buf ^ 1][(wK + i) * 64 + wN] = (__bf16)wreg[i];
    }
    wait_async0();
    __syncthreads();
    buf ^= 1;
  }

  int hi8 = (lane & 16) ? 8 : 0;
  int nCol0 = nBlock + (lane & 15);
#pragma unroll
  for (int half = 0; half < 2; ++half) {
    size_t mRow = mBlock + wave * 32 + half * 16 + hi8;
#pragma unroll
    for (int tt = 0; tt < 4; ++tt) {
      int nn = nCol0 + tt * 16;
      float bvv = bias[nn];
#pragma unroll
      for (int r = 0; r < 8; ++r) {
        size_t idx = (mRow + r) * Nout + nn;
        float val = acc[half * 4 + tt][r] + bvv;
        if (RELU) val = fmaxf(val, 0.0f);
        if (residual) val += residual[idx];
        if (outF) outF[idx] = val;
        if (outB) outB[idx] = (__bf16)val;
      }
    }
  }
}

// ---------------- fused flash attention: block = (n, head, 128 q-rows) -------
__global__ __launch_bounds__(256) void attn_kernel(
    const __bf16* __restrict__ Q, const __bf16* __restrict__ Kg,
    const __bf16* __restrict__ Vg, __bf16* __restrict__ O) {
  __shared__ __bf16 Qs[128 * 64];   // [m][dd]      16 KB
  __shared__ __bf16 KTs[64 * 64];   // [dd][s]       8 KB (transposed store)
  __shared__ __bf16 Vs[64 * 64];    // [s][dd]       8 KB
  __shared__ float Ss[128 * 64];    // fp32 scores  32 KB
  __shared__ __bf16 Ps[128 * 64];   // exp() bf16   16 KB
  __shared__ float mrow[128], lrow[128], crow[128];
  int tid = threadIdx.x;
  int lane = tid & 31, wave = tid >> 5;
  int hi8 = (lane & 16) ? 8 : 0;
  int qBase = blockIdx.x * 128;
  int head = blockIdx.y;
  size_t tokBase = (size_t)blockIdx.z * TSEQ;
  {  // async-stage Q tile (128x64), 64B per thread
    int r = tid >> 1, ddOff = (tid & 1) * 32;
    unsigned lds = lds_addr(&Qs[r * 64 + ddOff]);
    unsigned long long g = (unsigned long long)(size_t)(
        Q + (tokBase + qBase + r) * DMODEL + head * HDIM + ddOff);
    async_b128(lds, g);
    async_b128(lds + 16, g + 16);
    async_b128(lds + 32, g + 32);
    async_b128(lds + 48, g + 48);
  }
  if (tid < 128) {
    mrow[tid] = -1e30f;
    lrow[tid] = 0.0f;
  }
  v8f oacc[4] = {};
  int mB = wave * 16;
  for (int j = 0; j < TSEQ / 64; ++j) {
    __syncthreads();
    {  // stage K tile transposed (VALU scatter) + V tile (async copy)
      int s = tid >> 2, ddOff = (tid & 3) * 16;
      size_t tok = tokBase + j * 64 + s;
      const __bf16* kp = Kg + tok * DMODEL + head * HDIM + ddOff;
      __bf16 kv[16];
#pragma unroll
      for (int i = 0; i < 16; ++i) kv[i] = kp[i];
#pragma unroll
      for (int i = 0; i < 16; ++i) KTs[(ddOff + i) * 64 + s] = kv[i];
      unsigned lds = lds_addr(&Vs[s * 64 + ddOff]);
      unsigned long long g = (unsigned long long)(size_t)(
          Vg + tok * DMODEL + head * HDIM + ddOff);
      async_b128(lds, g);
      async_b128(lds + 16, g + 16);
    }
    wait_async0();
    __syncthreads();
    // S = Q @ K^T (128x64 per block, 16x64 per wave)
    v8f sacc[4] = {};
#pragma unroll
    for (int kk = 0; kk < HDIM; kk += 32) {
      v16bf a = frag_a_ld(Qs, 64, mB + (lane & 15), kk, lane);
#pragma unroll
      for (int tt = 0; tt < 4; ++tt) {
        v16bf b = frag_b_ld(KTs, 64, kk, tt * 16, lane);
        sacc[tt] = __builtin_amdgcn_wmma_f32_16x16x32_bf16(
            false, a, false, b, (short)0, sacc[tt], false, false);
      }
    }
#pragma unroll
    for (int tt = 0; tt < 4; ++tt)
#pragma unroll
      for (int r = 0; r < 8; ++r)
        Ss[(mB + r + hi8) * 64 + tt * 16 + (lane & 15)] = sacc[tt][r];
    __syncthreads();
    // online softmax (fp32), one thread per query row
    if (tid < 128) {
      int row = tid;
      float om = mrow[row];
      float tmax = -1e30f;
      for (int c = 0; c < 64; ++c)
        tmax = fmaxf(tmax, Ss[row * 64 + c] * 0.125f);  // 1/sqrt(64)
      float nm = fmaxf(om, tmax);
      float corr = __expf(om - nm);
      float sum = 0.0f;
      for (int c = 0; c < 64; ++c) {
        float p = __expf(Ss[row * 64 + c] * 0.125f - nm);
        Ps[row * 64 + c] = (__bf16)p;
        sum += p;
      }
      lrow[row] = lrow[row] * corr + sum;
      mrow[row] = nm;
      crow[row] = corr;
    }
    __syncthreads();
    // rescale running output, then O += P @ V
#pragma unroll
    for (int r = 0; r < 8; ++r) {
      float sc = crow[mB + r + hi8];
#pragma unroll
      for (int tt = 0; tt < 4; ++tt) oacc[tt][r] *= sc;
    }
#pragma unroll
    for (int kk = 0; kk < 64; kk += 32) {
      v16bf a = frag_a_ld(Ps, 64, mB + (lane & 15), kk, lane);
#pragma unroll
      for (int tt = 0; tt < 4; ++tt) {
        v16bf b = frag_b_ld(Vs, 64, kk, tt * 16, lane);
        oacc[tt] = __builtin_amdgcn_wmma_f32_16x16x32_bf16(
            false, a, false, b, (short)0, oacc[tt], false, false);
      }
    }
  }
#pragma unroll
  for (int tt = 0; tt < 4; ++tt)
#pragma unroll
    for (int r = 0; r < 8; ++r) {
      int row = mB + r + hi8;
      float val = oacc[tt][r] / lrow[row];
      O[(tokBase + qBase + row) * DMODEL + head * HDIM + tt * 16 +
        (lane & 15)] = (__bf16)val;
    }
}

// ---------------- kernel: X2 = Xn + proj; yn = LN(X2) bf16 -------------------
__global__ __launch_bounds__(256) void resid_ln_kernel(
    const float* __restrict__ Xn, const float* __restrict__ proj,
    const float* __restrict__ g, const float* __restrict__ be,
    float* __restrict__ X2, __bf16* __restrict__ yn) {
  __shared__ float red[256];
  size_t base = (size_t)blockIdx.x * DMODEL;
  int tid = threadIdx.x;
  float x[2];
#pragma unroll
  for (int i = 0; i < 2; ++i) {
    int c = tid + i * 256;
    x[i] = Xn[base + c] + proj[base + c];
    X2[base + c] = x[i];
  }
  float mean = block_sum(x[0] + x[1], red) * (1.0f / DMODEL);
  float d0 = x[0] - mean, d1 = x[1] - mean;
  float var = block_sum(d0 * d0 + d1 * d1, red) * (1.0f / (DMODEL - 1));
  float denom = sqrtf(var) + 1e-6f;
#pragma unroll
  for (int i = 0; i < 2; ++i) {
    int c = tid + i * 256;
    yn[base + c] = (__bf16)(g[c] * (x[i] - mean) / denom + be[c]);
  }
}

// ---------------- launcher ---------------------------------------------------
extern "C" void kernel_launch(void* const* d_in, const int* in_sizes, int n_in,
                              void* d_out, int out_size, void* d_ws,
                              size_t ws_size, hipStream_t stream) {
  const float* X = (const float*)d_in[0];
  const float* wq = (const float*)d_in[1];
  const float* bq = (const float*)d_in[2];
  const float* wk = (const float*)d_in[3];
  const float* bk = (const float*)d_in[4];
  const float* wv = (const float*)d_in[5];
  const float* bv = (const float*)d_in[6];
  const float* w1 = (const float*)d_in[7];
  const float* b1 = (const float*)d_in[8];
  const float* g_mha = (const float*)d_in[9];
  const float* be_mha = (const float*)d_in[10];
  const float* g_ffn = (const float*)d_in[11];
  const float* be_ffn = (const float*)d_in[12];
  const float* wf1 = (const float*)d_in[13];
  const float* bf1 = (const float*)d_in[14];
  const float* wf2 = (const float*)d_in[15];
  const float* bf2 = (const float*)d_in[16];

  const int N = in_sizes[0] / (TSEQ * DMODEL);
  const int M = N * TSEQ;  // 16384 tokens

  // workspace layout (liveness-based aliasing; peak = SF + 5*SB ~ 117 MB)
  char* ws = (char*)d_ws;
  const size_t SF = (size_t)M * DMODEL * sizeof(float);  // 33.5 MB
  const size_t SB = (size_t)M * DMODEL * 2;              // 16.8 MB
  float* Xn = (float*)(ws);                  // live: prep -> resid_ln
  __bf16* Xnb = (__bf16*)(ws + SF);          // live: prep -> v-gemm
  __bf16* qb = (__bf16*)(ws + SF + SB);      // live: -> attn
  __bf16* kb = (__bf16*)(ws + SF + 2 * SB);  // live: -> attn
  __bf16* vb = (__bf16*)(ws + SF + 3 * SB);  // live: -> attn
  __bf16* ao = (__bf16*)(ws + SF + 4 * SB);  // live: attn -> proj-gemm
  float* proj = (float*)(ws + SF);           // aliases Xnb+qb (dead by then)
  float* X2 = (float*)(ws + SF + 2 * SB);    // aliases kb+vb
  __bf16* yn = (__bf16*)(ws + SF + 4 * SB);  // aliases ao
  __bf16* hb = (__bf16*)(ws);                // aliases Xn (M x 1024 bf16)

  // 1) timing signal + LN
  prep_ln_kernel<<<M, 256, 0, stream>>>(X, g_mha, be_mha, Xn, Xnb);

  // 2-4) q/k/v = relu(Xn @ w.T + b)
  dim3 gD(M / 256, DMODEL / 64);
  gemm_kernel<true><<<gD, 256, 0, stream>>>(Xnb, wq, bq, nullptr, qb, nullptr,
                                            DMODEL, DMODEL);
  gemm_kernel<true><<<gD, 256, 0, stream>>>(Xnb, wk, bk, nullptr, kb, nullptr,
                                            DMODEL, DMODEL);
  gemm_kernel<true><<<gD, 256, 0, stream>>>(Xnb, wv, bv, nullptr, vb, nullptr,
                                            DMODEL, DMODEL);

  // 5) fused flash attention
  attn_kernel<<<dim3(TSEQ / 128, HEADS, N), 256, 0, stream>>>(qb, kb, vb, ao);

  // 6) out-projection, then residual + LN
  gemm_kernel<false><<<gD, 256, 0, stream>>>(ao, w1, b1, proj, nullptr,
                                             nullptr, DMODEL, DMODEL);
  resid_ln_kernel<<<M, 256, 0, stream>>>(Xn, proj, g_ffn, be_ffn, X2, yn);

  // 7) FFN: h = relu(yn @ wf1.T + bf1); out = X2 + h @ wf2.T + bf2
  gemm_kernel<true><<<dim3(M / 256, (2 * DMODEL) / 64), 256, 0, stream>>>(
      yn, wf1, bf1, nullptr, hb, nullptr, DMODEL, 2 * DMODEL);
  gemm_kernel<false><<<gD, 256, 0, stream>>>(hb, wf2, bf2, (float*)d_out,
                                             nullptr, X2, 2 * DMODEL, DMODEL);
}
